// HGNN_71270687310031
// MI455X (gfx1250) — matrix operand order repaired and checked by reading, not compiled, not run on hardware
//
#include <hip/hip_runtime.h>
#include <hip/hip_bf16.h>
#include <math.h>

typedef __attribute__((ext_vector_type(16))) _Float16 v16h;
typedef __attribute__((ext_vector_type(8)))  _Float16 v8h;
typedef __attribute__((ext_vector_type(8)))  float    v8f;

#define N_HEADS 2

// ---------------------------------------------------------------------------
// helpers
// ---------------------------------------------------------------------------
__device__ __forceinline__ unsigned ordEnc(float f) {
  unsigned u = __float_as_uint(f);
  return (u & 0x80000000u) ? ~u : (u | 0x80000000u);
}
__device__ __forceinline__ float ordDec(unsigned u) {
  return (u & 0x80000000u) ? __uint_as_float(u & 0x7FFFFFFFu) : __uint_as_float(~u);
}
__device__ __forceinline__ v16h ld_frag(const _Float16* p0, const _Float16* p1) {
  union { v8h h[2]; v16h v; } u;
  u.h[0] = *(const v8h*)p0;
  u.h[1] = *(const v8h*)p1;
  return u.v;
}

// ---------------------------------------------------------------------------
// elementwise kernels
// ---------------------------------------------------------------------------
__global__ void k_fill_f32(float* __restrict__ p, float v, long long n) {
  long long i = (long long)blockIdx.x * blockDim.x + threadIdx.x;
  if (i < n) p[i] = v;
}
__global__ void k_fill_u32(unsigned* __restrict__ p, unsigned v, long long n) {
  long long i = (long long)blockIdx.x * blockDim.x + threadIdx.x;
  if (i < n) p[i] = v;
}
__global__ void k_cvt_f16(const float* __restrict__ in, _Float16* __restrict__ out, long long n) {
  long long i = (long long)blockIdx.x * blockDim.x + threadIdx.x;
  if (i < n) out[i] = (_Float16)in[i];
}
__global__ void k_gelu_cvt_f16(const float* __restrict__ in, _Float16* __restrict__ out, long long n) {
  long long i = (long long)blockIdx.x * blockDim.x + threadIdx.x;
  if (i < n) {
    float x = in[i];
    out[i] = (_Float16)(0.5f * x * (1.0f + erff(x * 0.70710678118654752f)));
  }
}
// W is KxN row-major f32; WT becomes NxK row-major f16
__global__ void k_transpose_cvt(const float* __restrict__ W, _Float16* __restrict__ WT, int K, int N) {
  int t = blockIdx.x * blockDim.x + threadIdx.x;
  if (t >= K * N) return;
  int k = t / N, n = t - k * N;
  WT[(size_t)n * K + k] = (_Float16)W[t];
}

// ---------------------------------------------------------------------------
// WMMA GEMM: C[M,N] = A[M,K] (f16) * B[K,N] + bias, B given transposed (BT: NxK f16)
// Optional: sigmoid(gate)-weighted skip connection, ReLU.
// Block = 256 threads (8 waves); wave w -> rows [blk*128+16w, +16), cols [blkY*64, +64)
// N must be a multiple of 64; K a multiple of 32 (true for all call sites).
// ---------------------------------------------------------------------------
__global__ __launch_bounds__(256)
void k_wmma_gemm(const _Float16* __restrict__ A, int lda,
                 const _Float16* __restrict__ BT, int ldb,
                 const float* __restrict__ bias,
                 float* __restrict__ C, int ldc,
                 const float* __restrict__ skip, int ldskip,
                 const float* __restrict__ gate,
                 int M, int N, int K, int relu) {
  const int wave = threadIdx.x >> 5;
  const int lane = threadIdx.x & 31;
  const int hi   = lane >> 4;
  const int l16  = lane & 15;

  const int rowBase = blockIdx.x * 128 + wave * 16;
  const int colBase = blockIdx.y * 64;

  int mrow  = rowBase + l16;
  int mload = mrow < M ? mrow : (M - 1);

  const _Float16* Arow = A + (size_t)mload * lda;
  const _Float16* Brow0 = BT + (size_t)(colBase +  0 + l16) * ldb;
  const _Float16* Brow1 = BT + (size_t)(colBase + 16 + l16) * ldb;
  const _Float16* Brow2 = BT + (size_t)(colBase + 32 + l16) * ldb;
  const _Float16* Brow3 = BT + (size_t)(colBase + 48 + l16) * ldb;

  v8f acc0 = {}, acc1 = {}, acc2 = {}, acc3 = {};

  for (int k0 = 0; k0 < K; k0 += 32) {
    // A fragment (16x32): element e holds K = e + 16*hi
    v16h a = ld_frag(Arow + k0 + 16 * hi, Arow + k0 + 16 * hi + 8);
    // B fragment (32x16): element e holds K = e + 16*hi
    v16h b0 = ld_frag(Brow0 + k0 + 16 * hi, Brow0 + k0 + 16 * hi + 8);
    v16h b1 = ld_frag(Brow1 + k0 + 16 * hi, Brow1 + k0 + 16 * hi + 8);
    v16h b2 = ld_frag(Brow2 + k0 + 16 * hi, Brow2 + k0 + 16 * hi + 8);
    v16h b3 = ld_frag(Brow3 + k0 + 16 * hi, Brow3 + k0 + 16 * hi + 8);
    acc0 = __builtin_amdgcn_wmma_f32_16x16x32_f16(false, a, false, b0, (short)0, acc0, false, false);
    acc1 = __builtin_amdgcn_wmma_f32_16x16x32_f16(false, a, false, b1, (short)0, acc1, false, false);
    acc2 = __builtin_amdgcn_wmma_f32_16x16x32_f16(false, a, false, b2, (short)0, acc2, false, false);
    acc3 = __builtin_amdgcn_wmma_f32_16x16x32_f16(false, a, false, b3, (short)0, acc3, false, false);
  }

  float g = 0.0f;
  if (skip) g = 1.0f / (1.0f + expf(-gate[0]));

  v8f accs[4] = {acc0, acc1, acc2, acc3};
#pragma unroll
  for (int t = 0; t < 4; ++t) {
    int col = colBase + 16 * t + l16;
    float bv = bias ? bias[col] : 0.0f;
#pragma unroll
    for (int r = 0; r < 8; ++r) {
      int m = rowBase + r + 8 * hi;   // C/D layout: VGPR r <-> M = r + 8*hi
      if (m < M) {
        float v = accs[t][r] + bv;
        if (skip) v = g * v + (1.0f - g) * skip[(size_t)m * ldskip + col];
        if (relu) v = v > 0.0f ? v : 0.0f;
        C[(size_t)m * ldc + col] = v;
      }
    }
  }
}

// ---------------------------------------------------------------------------
// edge kernels (one (edge, head) per thread)
// ---------------------------------------------------------------------------
__global__ void k_edge_logits(const int* __restrict__ src, const int* __restrict__ dst,
                              const float* __restrict__ Q, const float* __restrict__ Kr,
                              const float* __restrict__ prel,
                              float* __restrict__ ew, unsigned* __restrict__ maxb,
                              int E, int D, int outc) {
  int t = blockIdx.x * blockDim.x + threadIdx.x;
  if (t >= E * N_HEADS) return;
  int i = t >> 1, h = t & 1;
  int s0 = src[i], d0 = dst[i];
  const float* q = Q + (size_t)d0 * outc + h * D;
  const float* k = Kr + (size_t)s0 * outc + h * D;
  float acc = 0.0f;
  for (int j = 0; j < D; j += 4) {
    float4 qv = *(const float4*)(q + j);
    float4 kv = *(const float4*)(k + j);
    acc += qv.x * kv.x + qv.y * kv.y + qv.z * kv.z + qv.w * kv.w;
  }
  float logit = acc * prel[h] * rsqrtf((float)D);
  ew[t] = logit;
  atomicMax(maxb + (size_t)d0 * N_HEADS + h, ordEnc(logit));
}

__global__ void k_edge_expsum(const int* __restrict__ dst, float* __restrict__ ew,
                              const unsigned* __restrict__ maxb, float* __restrict__ sumb, int E) {
  int t = blockIdx.x * blockDim.x + threadIdx.x;
  if (t >= E * N_HEADS) return;
  int i = t >> 1, h = t & 1;
  int d0 = dst[i];
  float m = ordDec(maxb[(size_t)d0 * N_HEADS + h]);
  if (!isfinite(m)) m = 0.0f;
  float e = expf(ew[t] - m);
  ew[t] = e;
  atomicAdd(sumb + (size_t)d0 * N_HEADS + h, e);
}

__global__ void k_edge_msg(const int* __restrict__ src, const int* __restrict__ dst,
                           const float* __restrict__ Vr, const float* __restrict__ ew,
                           const float* __restrict__ sumb, float* __restrict__ agg,
                           long long total, int D, int outc) {
  long long t = (long long)blockIdx.x * blockDim.x + threadIdx.x;
  if (t >= total) return;
  int c = (int)(t % outc);
  long long i = t / outc;
  int h = c / D;
  int s0 = src[i], d0 = dst[i];
  float alpha = ew[i * N_HEADS + h] / sumb[(size_t)d0 * N_HEADS + h];
  atomicAdd(agg + (size_t)d0 * outc + c, Vr[(size_t)s0 * outc + c] * alpha);
}

// ---------------------------------------------------------------------------
// host-side launch helpers
// ---------------------------------------------------------------------------
static inline long long cdiv(long long a, long long b) { return (a + b - 1) / b; }

static void gemm(hipStream_t s, const _Float16* A, int lda, const _Float16* BT, int ldb,
                 const float* bias, float* C, int ldc,
                 const float* skip, int ldskip, const float* gate,
                 int M, int N, int K, int relu) {
  dim3 grid((unsigned)cdiv(M, 128), (unsigned)(N / 64));
  k_wmma_gemm<<<grid, 256, 0, s>>>(A, lda, BT, ldb, bias, C, ldc, skip, ldskip, gate, M, N, K, relu);
}
static void cvt16(hipStream_t s, const float* in, _Float16* out, long long n) {
  k_cvt_f16<<<(unsigned)cdiv(n, 256), 256, 0, s>>>(in, out, n);
}
static void gelu16(hipStream_t s, const float* in, _Float16* out, long long n) {
  k_gelu_cvt_f16<<<(unsigned)cdiv(n, 256), 256, 0, s>>>(in, out, n);
}
static void fillf(hipStream_t s, float* p, float v, long long n) {
  k_fill_f32<<<(unsigned)cdiv(n, 256), 256, 0, s>>>(p, v, n);
}
static void fillu(hipStream_t s, unsigned* p, unsigned v, long long n) {
  k_fill_u32<<<(unsigned)cdiv(n, 256), 256, 0, s>>>(p, v, n);
}
static void trw(hipStream_t s, const float* W, _Float16* WT, int K, int N) {
  k_transpose_cvt<<<(unsigned)cdiv((long long)K * N, 256), 256, 0, s>>>(W, WT, K, N);
}

// ---------------------------------------------------------------------------
// kernel_launch
// ---------------------------------------------------------------------------
extern "C" void kernel_launch(void* const* d_in, const int* in_sizes, int n_in,
                              void* d_out, int out_size, void* d_ws, size_t ws_size,
                              hipStream_t stream) {
  (void)n_in; (void)out_size; (void)ws_size;
  const int Nn = in_sizes[0] / 64;     // 50000
  const int Na = in_sizes[1] / 64;     // 10000

  const float* x_node  = (const float*)d_in[0];
  const float* x_agent = (const float*)d_in[1];
  const int*   ei[3]   = {(const int*)d_in[87], (const int*)d_in[88], (const int*)d_in[89]};
  const int    Ecnt[3] = {in_sizes[87] / 2, in_sizes[88] / 2, in_sizes[89] / 2};

  // -------- workspace carve-out (all 256B aligned) --------
  char* wptr = (char*)d_ws;
  auto take = [&](size_t bytes) -> void* {
    void* p = (void*)wptr;
    wptr += (bytes + 255) & ~(size_t)255;
    return p;
  };
  float*    hN0   = (float*)take((size_t)Nn * 128 * 4);
  float*    hN1   = (float*)take((size_t)Nn * 128 * 4);
  float*    hA0   = (float*)take((size_t)Na * 128 * 4);
  float*    hA1   = (float*)take((size_t)Na * 128 * 4);
  _Float16* h16N  = (_Float16*)take((size_t)Nn * 128 * 2);
  _Float16* h16A  = (_Float16*)take((size_t)Na * 128 * 2);
  float*    QN    = (float*)take((size_t)Nn * 256 * 4);
  float*    QA    = (float*)take((size_t)Na * 256 * 4);
  _Float16* KN16  = (_Float16*)take((size_t)Nn * 256 * 2);
  _Float16* VN16  = (_Float16*)take((size_t)Nn * 256 * 2);
  _Float16* KA16  = (_Float16*)take((size_t)Na * 256 * 2);
  _Float16* VA16  = (_Float16*)take((size_t)Na * 256 * 2);
  float*    krel  = (float*)take((size_t)Nn * 256 * 4);   // also f32 GEMM staging
  float*    vrel  = (float*)take((size_t)Nn * 256 * 4);
  float*    aggN  = (float*)take((size_t)Nn * 256 * 4);
  float*    aggA  = (float*)take((size_t)Na * 256 * 4);
  _Float16* g16N  = (_Float16*)take((size_t)Nn * 256 * 2);
  _Float16* g16A  = (_Float16*)take((size_t)Na * 256 * 2);
  unsigned* maxb  = (unsigned*)take((size_t)Nn * 2 * 4);
  float*    sumb  = (float*)take((size_t)Nn * 2 * 4);
  float*    ew    = (float*)take((size_t)Ecnt[0] * 2 * 4);
  _Float16* warena = (_Float16*)take((size_t)8 * 1024 * 1024);

  const unsigned NEG_INF_ORD = 0x007FFFFFu;  // ordEnc(-inf)

  // ================= front MLP: h = relu(relu(x@W1+b1)@W2+b2) =================
  {
    _Float16* wp = warena;
    auto wtake = [&](size_t halfs) { _Float16* p = wp; wp += (halfs + 127) & ~(size_t)127; return p; };
    _Float16* W1T = wtake(64 * 128);  trw(stream, (const float*)d_in[2], W1T, 64, 128);
    _Float16* W2T = wtake(128 * 128); trw(stream, (const float*)d_in[4], W2T, 128, 128);
    const float* b1 = (const float*)d_in[3];
    const float* b2 = (const float*)d_in[5];

    cvt16(stream, x_node, h16N, (long long)Nn * 64);
    gemm(stream, h16N, 64, W1T, 64, b1, krel, 128, nullptr, 0, nullptr, Nn, 128, 64, 1);
    cvt16(stream, krel, KN16, (long long)Nn * 128);
    gemm(stream, KN16, 128, W2T, 128, b2, hN0, 128, nullptr, 0, nullptr, Nn, 128, 128, 1);

    cvt16(stream, x_agent, h16A, (long long)Na * 64);
    gemm(stream, h16A, 64, W1T, 64, b1, krel, 128, nullptr, 0, nullptr, Na, 128, 64, 1);
    cvt16(stream, krel, KA16, (long long)Na * 128);
    gemm(stream, KA16, 128, W2T, 128, b2, hA0, 128, nullptr, 0, nullptr, Na, 128, 128, 1);
  }

  // ================= three HGT conv layers =================
  const int emSrc[3] = {0, 1, 0};  // nn: node->node, an: agent->node, na: node->agent
  const int emDst[3] = {0, 0, 1};

  for (int layer = 0; layer < 3; ++layer) {
    const int base  = 6 + layer * 27;
    const int out_c = (layer == 2) ? 256 : 128;
    const int D     = out_c / N_HEADS;
    const bool last = (layer == 2);

    const float* hNin = (layer == 1) ? hN1 : hN0;
    const float* hAin = (layer == 1) ? hA1 : hA0;
    float* hNout = (layer == 0) ? hN1 : (layer == 1 ? hN0 : (float*)d_out);
    float* hAout = (layer == 0) ? hA1 : hA0;

    // ---- transpose+convert all weights for this layer ----
    _Float16* wp = warena;
    auto wtake = [&](size_t halfs) { _Float16* p = wp; wp += (halfs + 127) & ~(size_t)127; return p; };
    _Float16 *WkT[2], *WqT[2], *WvT[2], *WaT[2], *AT[3][2], *MT[3][2];
    for (int t = 0; t < 2; ++t) {
      int b = base + t * 9;
      WkT[t] = wtake((size_t)out_c * 128);  trw(stream, (const float*)d_in[b + 0], WkT[t], 128, out_c);
      WqT[t] = wtake((size_t)out_c * 128);  trw(stream, (const float*)d_in[b + 2], WqT[t], 128, out_c);
      WvT[t] = wtake((size_t)out_c * 128);  trw(stream, (const float*)d_in[b + 4], WvT[t], 128, out_c);
      WaT[t] = wtake((size_t)out_c * out_c); trw(stream, (const float*)d_in[b + 6], WaT[t], out_c, out_c);
    }
    for (int e = 0; e < 3; ++e) {
      int b = base + 18 + e * 3;
      for (int h = 0; h < 2; ++h) {
        AT[e][h] = wtake((size_t)D * D); trw(stream, (const float*)d_in[b + 0] + (size_t)h * D * D, AT[e][h], D, D);
        MT[e][h] = wtake((size_t)D * D); trw(stream, (const float*)d_in[b + 1] + (size_t)h * D * D, MT[e][h], D, D);
      }
    }

    // ---- convert inputs to f16 ----
    cvt16(stream, hNin, h16N, (long long)Nn * 128);
    cvt16(stream, hAin, h16A, (long long)Na * 128);

    // ---- Q / K / V projections ----
    gemm(stream, h16N, 128, WqT[0], 128, (const float*)d_in[base + 3], QN, out_c,
         nullptr, 0, nullptr, Nn, out_c, 128, 0);
    if (!last)
      gemm(stream, h16A, 128, WqT[1], 128, (const float*)d_in[base + 12], QA, out_c,
           nullptr, 0, nullptr, Na, out_c, 128, 0);

    gemm(stream, h16N, 128, WkT[0], 128, (const float*)d_in[base + 1], krel, out_c,
         nullptr, 0, nullptr, Nn, out_c, 128, 0);
    cvt16(stream, krel, KN16, (long long)Nn * out_c);
    gemm(stream, h16N, 128, WvT[0], 128, (const float*)d_in[base + 5], vrel, out_c,
         nullptr, 0, nullptr, Nn, out_c, 128, 0);
    cvt16(stream, vrel, VN16, (long long)Nn * out_c);
    gemm(stream, h16A, 128, WkT[1], 128, (const float*)d_in[base + 10], krel, out_c,
         nullptr, 0, nullptr, Na, out_c, 128, 0);
    cvt16(stream, krel, KA16, (long long)Na * out_c);
    gemm(stream, h16A, 128, WvT[1], 128, (const float*)d_in[base + 14], vrel, out_c,
         nullptr, 0, nullptr, Na, out_c, 128, 0);
    cvt16(stream, vrel, VA16, (long long)Na * out_c);

    // ---- zero aggregation buffers ----
    fillf(stream, aggN, 0.0f, (long long)Nn * out_c);
    if (!last) fillf(stream, aggA, 0.0f, (long long)Na * out_c);

    // ---- per-edge-type attention + aggregation ----
    for (int e = 0; e < 3; ++e) {
      if (last && e == 2) continue;  // 'na' dst=agent unused in final layer
      const int E = Ecnt[e];
      const int* srcI = ei[e];
      const int* dstI = ei[e] + E;
      const _Float16* K16s = emSrc[e] ? KA16 : KN16;
      const _Float16* V16s = emSrc[e] ? VA16 : VN16;
      const int Nsrc = emSrc[e] ? Na : Nn;
      const int Ndst = emDst[e] ? Na : Nn;
      const float* Qd  = emDst[e] ? QA : QN;
      float* aggd      = emDst[e] ? aggA : aggN;
      const float* prel = (const float*)d_in[base + 18 + e * 3 + 2];

      // per-head relation transforms: k = K_src @ a_rel[h], v = V_src @ m_rel[h]
      for (int h = 0; h < 2; ++h) {
        gemm(stream, K16s + h * D, out_c, AT[e][h], D, nullptr, krel + h * D, out_c,
             nullptr, 0, nullptr, Nsrc, D, D, 0);
        gemm(stream, V16s + h * D, out_c, MT[e][h], D, nullptr, vrel + h * D, out_c,
             nullptr, 0, nullptr, Nsrc, D, D, 0);
      }

      fillu(stream, maxb, NEG_INF_ORD, (long long)Ndst * 2);
      fillf(stream, sumb, 0.0f, (long long)Ndst * 2);

      k_edge_logits<<<(unsigned)cdiv((long long)E * 2, 256), 256, 0, stream>>>(
          srcI, dstI, Qd, krel, prel, ew, maxb, E, D, out_c);
      k_edge_expsum<<<(unsigned)cdiv((long long)E * 2, 256), 256, 0, stream>>>(
          dstI, ew, maxb, sumb, E);
      long long tot = (long long)E * out_c;
      k_edge_msg<<<(unsigned)cdiv(tot, 256), 256, 0, stream>>>(
          srcI, dstI, vrel, ew, sumb, aggd, tot, D, out_c);
    }

    // ---- output projection: relu( gate-skip( gelu(agg) @ Wa + ba ) ) ----
    {
      gelu16(stream, aggN, g16N, (long long)Nn * out_c);
      const float* skipN = (out_c == 128) ? hNin : nullptr;
      gemm(stream, g16N, out_c, WaT[0], out_c, (const float*)d_in[base + 7], hNout, out_c,
           skipN, 128, (const float*)d_in[base + 8], Nn, out_c, out_c, 1);
      if (!last) {
        gelu16(stream, aggA, g16A, (long long)Na * out_c);
        gemm(stream, g16A, out_c, WaT[1], out_c, (const float*)d_in[base + 16], hAout, out_c,
             hAin, 128, (const float*)d_in[base + 17], Na, out_c, out_c, 1);
      }
    }
  }
}
